// RoIHead_39951785787755
// MI455X (gfx1250) — matrix operand order repaired and checked by reading, not compiled, not run on hardware
//
#include <hip/hip_runtime.h>
#include <hip/hip_bf16.h>
#include <math.h>

// ---------- problem constants ----------
#define IN_CH 256
#define NUM_CLASSES 21
#define ROI 7
#define SRR 2
#define BB_ 2
#define NPROP 128
#define FH 32
#define FW 32
#define NROI (BB_ * NPROP)           // 256
#define FEAT_DIM (IN_CH * ROI * ROI) // 12544
#define HID 1024
#define MAX_DET 100
#define NCB (NPROP * (NUM_CLASSES - 1)) // 2560 class-boxes per image

// ---------- CDNA5 vector types ----------
typedef __attribute__((ext_vector_type(16))) __bf16 v16bf;
typedef __attribute__((ext_vector_type(8)))  float  v8f;
typedef __attribute__((ext_vector_type(4)))  float  v4f;
typedef __attribute__((ext_vector_type(4)))  unsigned int v4u;
typedef __attribute__((ext_vector_type(2)))  unsigned int v2u;

union Frag { v4u q[2]; v16bf v; };

// fp32 -> bf16 hi/lo split (RNE), hi+lo captures ~16 mantissa bits.
__device__ __forceinline__ void bfsplit(float f, unsigned short& h, unsigned short& l) {
  unsigned u = __float_as_uint(f);
  unsigned r = u + 0x7FFFu + ((u >> 16) & 1u);
  h = (unsigned short)(r >> 16);
  float hf = __uint_as_float(((unsigned)h) << 16);
  float lo = f - hf;
  unsigned ul = __float_as_uint(lo);
  unsigned rl = ul + 0x7FFFu + ((ul >> 16) & 1u);
  l = (unsigned short)(rl >> 16);
}

// ============================================================
// 1) RoIAlign: one block per roi (256 threads = 256 channels)
// ============================================================
__global__ __launch_bounds__(256) void roi_align_kernel(
    const float* __restrict__ features,   // [B,256,32,32]
    const float* __restrict__ proposals,  // [B,128,4] flat
    float* __restrict__ feats)            // [256, 12544] channel-major flatten
{
  const int r = blockIdx.x;
  const int c = threadIdx.x;
  const int b = r / NPROP;
  const float* pr = proposals + (size_t)r * 4;
  const float scale = 1.0f / 32.0f;
  float x1 = pr[0] * scale, y1 = pr[1] * scale;
  float x2 = pr[2] * scale, y2 = pr[3] * scale;
  float rw = fmaxf(x2 - x1, 1.0f), rh = fmaxf(y2 - y1, 1.0f);
  float bw = rw / ROI, bh = rh / ROI;
  const float* F = features + ((size_t)b * IN_CH + c) * (FH * FW);

  for (int py = 0; py < ROI; ++py) {
    for (int px = 0; px < ROI; ++px) {
      float acc = 0.0f;
      #pragma unroll
      for (int sy = 0; sy < SRR; ++sy) {
        #pragma unroll
        for (int sx = 0; sx < SRR; ++sx) {
          float y = y1 + ((float)py + ((float)sy + 0.5f) / SRR) * bh;
          float x = x1 + ((float)px + ((float)sx + 0.5f) / SRR) * bw;
          bool inb = (y >= -1.0f) && (y <= (float)FH) && (x >= -1.0f) && (x <= (float)FW);
          float yc = fminf(fmaxf(y, 0.0f), (float)(FH - 1));
          float xc = fminf(fmaxf(x, 0.0f), (float)(FW - 1));
          float y0f = floorf(yc), x0f = floorf(xc);
          int y0 = (int)y0f, x0 = (int)x0f;
          int y1i = min(y0 + 1, FH - 1), x1i = min(x0 + 1, FW - 1);
          float ly = yc - y0f, lx = xc - x0f;
          float hy = 1.0f - ly, hx = 1.0f - lx;
          float f00 = F[y0 * FW + x0],  f01 = F[y0 * FW + x1i];
          float f10 = F[y1i * FW + x0], f11 = F[y1i * FW + x1i];
          float v = hy * hx * f00 + hy * lx * f01 + ly * hx * f10 + ly * lx * f11;
          acc += inb ? v : 0.0f;
        }
      }
      feats[(size_t)r * FEAT_DIM + c * (ROI * ROI) + py * ROI + px] = acc * 0.25f;
    }
  }
}

// ============================================================
// 2) Split-K GEMM partials: P[z][N][M] = A[:,kz] @ W[:,kz]^T
//    fp32 streamed once, bf16 hi/lo split in-register,
//    3x v_wmma_f32_16x16x32_bf16 per tile. Double-buffered LDS,
//    one barrier per K-step; grid.z = K-split for HBM MLP.
// ============================================================
#define TN 64
#define TM 128
#define TK 32

__global__ __launch_bounds__(256) void gemm_bf16x3_splitk_kernel(
    const float* __restrict__ A, const float* __restrict__ W,
    float* __restrict__ P,               // [gridDim.z][NROI][M] partials
    int K, int M, int kchunk)
{
  __shared__ unsigned short Ah[2][TN * TK], Al[2][TN * TK];
  __shared__ unsigned short Bh[2][TM * TK], Bl[2][TM * TK];

  const int tid  = threadIdx.x;
  const int lane = tid & 31;
  const int wave = tid >> 5;            // wave32
  const int nsub = wave & 3;
  const int mgrp = wave >> 2;
  const int n0 = blockIdx.y * TN;
  const int m0 = blockIdx.x * TM;
  const int k0 = blockIdx.z * kchunk;
  const int nkc = kchunk / TK;

  v8f acc[4];
  const v8f z = {0.f,0.f,0.f,0.f,0.f,0.f,0.f,0.f};
  #pragma unroll
  for (int j = 0; j < 4; ++j) acc[j] = z;

  // LDS frag addresses matching ISA 16-bit A(16x32)/B(32x16) layouts.
  const int arow = nsub * 16 + (lane & 15);
  const int aoff = arow * TK + ((lane & 16) ? 8 : 0);
  const int brow_lane = lane & 15;
  const int boff_k = (lane & 16) ? 8 : 0;

  v4f ra[2], rb[4];

  auto gload = [&](int kt) {
    #pragma unroll
    for (int rep = 0; rep < 2; ++rep) {
      int id = tid + rep * 256;
      int row = id >> 3, c4 = (id & 7) << 2;
      ra[rep] = *(const v4f*)&A[(size_t)(n0 + row) * K + k0 + kt * TK + c4];
    }
    #pragma unroll
    for (int rep = 0; rep < 4; ++rep) {
      int id = tid + rep * 256;
      int row = id >> 3, c4 = (id & 7) << 2;
      rb[rep] = *(const v4f*)&W[(size_t)(m0 + row) * K + k0 + kt * TK + c4];
    }
  };

  auto lstore = [&](int buf) {
    #pragma unroll
    for (int rep = 0; rep < 2; ++rep) {
      int id = tid + rep * 256;
      int row = id >> 3, c4 = (id & 7) << 2;
      unsigned short h0,h1,h2,h3,l0,l1,l2,l3;
      bfsplit(ra[rep].x,h0,l0); bfsplit(ra[rep].y,h1,l1);
      bfsplit(ra[rep].z,h2,l2); bfsplit(ra[rep].w,h3,l3);
      *(v2u*)&Ah[buf][row * TK + c4] = (v2u){((unsigned)h1<<16)|h0, ((unsigned)h3<<16)|h2};
      *(v2u*)&Al[buf][row * TK + c4] = (v2u){((unsigned)l1<<16)|l0, ((unsigned)l3<<16)|l2};
    }
    #pragma unroll
    for (int rep = 0; rep < 4; ++rep) {
      int id = tid + rep * 256;
      int row = id >> 3, c4 = (id & 7) << 2;
      unsigned short h0,h1,h2,h3,l0,l1,l2,l3;
      bfsplit(rb[rep].x,h0,l0); bfsplit(rb[rep].y,h1,l1);
      bfsplit(rb[rep].z,h2,l2); bfsplit(rb[rep].w,h3,l3);
      *(v2u*)&Bh[buf][row * TK + c4] = (v2u){((unsigned)h1<<16)|h0, ((unsigned)h3<<16)|h2};
      *(v2u*)&Bl[buf][row * TK + c4] = (v2u){((unsigned)l1<<16)|l0, ((unsigned)l3<<16)|l2};
    }
  };

  // prologue: stage K-step 0
  gload(0);
  lstore(0);
  __syncthreads();

  for (int kt = 0; kt < nkc; ++kt) {
    const int cur = kt & 1;
    const bool more = (kt + 1 < nkc);
    if (more) gload(kt + 1);                 // overlap with WMMA below
    if (kt + 2 < nkc) {                      // global_prefetch_b8 for kt+2
      __builtin_prefetch(&A[(size_t)(n0 + (tid >> 3)) * K + k0 + (kt + 2) * TK], 0, 0);
      __builtin_prefetch(&W[(size_t)(m0 + (tid >> 1)) * K + k0 + (kt + 2) * TK], 0, 0);
    }

    Frag a_h, a_l;
    a_h.q[0] = *(const v4u*)&Ah[cur][aoff];
    a_h.q[1] = *(const v4u*)&Ah[cur][aoff + 16];
    a_l.q[0] = *(const v4u*)&Al[cur][aoff];
    a_l.q[1] = *(const v4u*)&Al[cur][aoff + 16];

    #pragma unroll
    for (int j = 0; j < 4; ++j) {
      int msub = mgrp * 4 + j;
      int boff = (msub * 16 + brow_lane) * TK + boff_k;
      Frag b_h, b_l;
      b_h.q[0] = *(const v4u*)&Bh[cur][boff];
      b_h.q[1] = *(const v4u*)&Bh[cur][boff + 16];
      b_l.q[0] = *(const v4u*)&Bl[cur][boff];
      b_l.q[1] = *(const v4u*)&Bl[cur][boff + 16];
      // (ah+al)*(bh+bl) ~= ah*bh + ah*bl + al*bh
      acc[j] = __builtin_amdgcn_wmma_f32_16x16x32_bf16(false, a_h.v, false, b_h.v,
                                                       (short)0, acc[j], false, false);
      acc[j] = __builtin_amdgcn_wmma_f32_16x16x32_bf16(false, a_h.v, false, b_l.v,
                                                       (short)0, acc[j], false, false);
      acc[j] = __builtin_amdgcn_wmma_f32_16x16x32_bf16(false, a_l.v, false, b_h.v,
                                                       (short)0, acc[j], false, false);
    }

    if (more) lstore((kt + 1) & 1);          // write other buffer: no conflict
    __syncthreads();                          // one barrier per K-step
  }

  // epilogue -> fp32 partials (bias/relu applied in reduce kernel)
  float* Pz = P + (size_t)blockIdx.z * NROI * M;
  #pragma unroll
  for (int j = 0; j < 4; ++j) {
    int msub = mgrp * 4 + j;
    int m = m0 + msub * 16 + (lane & 15);
    #pragma unroll
    for (int r = 0; r < 8; ++r) {
      int n = n0 + nsub * 16 + r + ((lane & 16) ? 8 : 0);
      Pz[(size_t)n * M + m] = acc[j][r];
    }
  }
}

// ---- reduce split-K partials, add bias, ReLU ----
__global__ __launch_bounds__(256) void reduce_bias_relu_kernel(
    const float* __restrict__ P, const float* __restrict__ bias,
    float* __restrict__ C, int M, int nsplit)
{
  int i = blockIdx.x * blockDim.x + threadIdx.x;   // over N*M/4
  size_t e = (size_t)i * 4;
  v4f s = *(const v4f*)&P[e];
  for (int zz = 1; zz < nsplit; ++zz) {
    v4f p = *(const v4f*)&P[(size_t)zz * NROI * M + e];
    s.x += p.x; s.y += p.y; s.z += p.z; s.w += p.w;
  }
  int m = (int)(e % (size_t)M);
  v4f b = *(const v4f*)&bias[m];
  v4f o;
  o.x = fmaxf(s.x + b.x, 0.0f);
  o.y = fmaxf(s.y + b.y, 0.0f);
  o.z = fmaxf(s.z + b.z, 0.0f);
  o.w = fmaxf(s.w + b.w, 0.0f);
  *(v4f*)&C[e] = o;
}

// ============================================================
// 3) Heads: cls logits + softmax, box regression. Block per roi.
// ============================================================
__global__ __launch_bounds__(128) void heads_kernel(
    const float* __restrict__ X,
    const float* __restrict__ Wc, const float* __restrict__ bc,
    const float* __restrict__ Wb, const float* __restrict__ bb,
    float* __restrict__ probs, float* __restrict__ reg)
{
  __shared__ float xv[HID];
  __shared__ float lg[NUM_CLASSES];
  __shared__ float red[2];
  const int r = blockIdx.x, t = threadIdx.x;
  for (int k = t; k < HID; k += 128) xv[k] = X[(size_t)r * HID + k];
  __syncthreads();

  if (t < NUM_CLASSES + NUM_CLASSES * 4) {
    const float* wrow;
    float bias;
    if (t < NUM_CLASSES) { wrow = Wc + (size_t)t * HID; bias = bc[t]; }
    else { wrow = Wb + (size_t)(t - NUM_CLASSES) * HID; bias = bb[t - NUM_CLASSES]; }
    float s = bias;
    for (int k = 0; k < HID; ++k) s += xv[k] * wrow[k];
    if (t < NUM_CLASSES) lg[t] = s;
    else reg[(size_t)r * (NUM_CLASSES * 4) + (t - NUM_CLASSES)] = s;
  }
  __syncthreads();
  if (t == 0) {
    float m = lg[0];
    for (int j = 1; j < NUM_CLASSES; ++j) m = fmaxf(m, lg[j]);
    float s = 0.0f;
    for (int j = 0; j < NUM_CLASSES; ++j) s += expf(lg[j] - m);
    red[0] = m; red[1] = s;
  }
  __syncthreads();
  if (t < NUM_CLASSES)
    probs[(size_t)r * NUM_CLASSES + t] = expf(lg[t] - red[0]) / red[1];
}

// ============================================================
// 4) Postprocess: decode + clip + class offset + greedy NMS.
// ============================================================
__global__ __launch_bounds__(256) void post_kernel(
    const float* __restrict__ probs, const float* __restrict__ reg,
    const float* __restrict__ proposals, const int* __restrict__ img_sizes,
    float* __restrict__ scratch, float* __restrict__ out)
{
  const int i = blockIdx.x, tid = threadIdx.x;
  float* boxes = scratch + (size_t)i * 30720;
  float* boff  = boxes + NCB * 4;
  float* sc    = boff + NCB * 4;
  float* area  = sc + NCB;
  int*   lab   = (int*)(area + NCB);
  int*   rem   = lab + NCB;

  __shared__ float s_sc[256];
  __shared__ int   s_si[256];
  __shared__ float s_misc[1];

  const float Wimg = (float)img_sizes[i * 2 + 1];
  const float Himg = (float)img_sizes[i * 2 + 0];

  for (int idx = tid; idx < NCB; idx += 256) {
    int p = idx / (NUM_CLASSES - 1);
    int c = idx % (NUM_CLASSES - 1) + 1;
    const float* pr = proposals + ((size_t)i * NPROP + p) * 4;
    float pw = pr[2] - pr[0], ph = pr[3] - pr[1];
    float pcx = pr[0] + pw * 0.5f, pcy = pr[1] + ph * 0.5f;
    const float* d = reg + ((size_t)i * NPROP + p) * (NUM_CLASSES * 4) + c * 4;
    float cx = pcx + d[0] * pw, cy = pcy + d[1] * ph;
    float w = pw * expf(d[2]), h = ph * expf(d[3]);
    float x1 = fminf(fmaxf(cx - w * 0.5f, 0.f), Wimg);
    float y1 = fminf(fmaxf(cy - h * 0.5f, 0.f), Himg);
    float x2 = fminf(fmaxf(cx + w * 0.5f, 0.f), Wimg);
    float y2 = fminf(fmaxf(cy + h * 0.5f, 0.f), Himg);
    float s = probs[((size_t)i * NPROP + p) * NUM_CLASSES + c];
    boxes[idx * 4 + 0] = x1; boxes[idx * 4 + 1] = y1;
    boxes[idx * 4 + 2] = x2; boxes[idx * 4 + 3] = y2;
    sc[idx] = s; lab[idx] = c;
    rem[idx] = (s > 0.05f) ? 0 : 1;
  }
  __syncthreads();

  float lm = 0.0f;
  for (int idx = tid; idx < NCB; idx += 256)
    if (!rem[idx]) {
      float m4 = fmaxf(fmaxf(boxes[idx*4+0], boxes[idx*4+1]),
                       fmaxf(boxes[idx*4+2], boxes[idx*4+3]));
      lm = fmaxf(lm, m4);
    }
  s_sc[tid] = lm; __syncthreads();
  for (int o = 128; o > 0; o >>= 1) {
    if (tid < o) s_sc[tid] = fmaxf(s_sc[tid], s_sc[tid + o]);
    __syncthreads();
  }
  if (tid == 0) s_misc[0] = s_sc[0];
  __syncthreads();
  const float off_unit = s_misc[0] + 1.0f;

  for (int idx = tid; idx < NCB; idx += 256) {
    float off = (float)lab[idx] * off_unit;
    float ox1 = boxes[idx*4+0] + off, oy1 = boxes[idx*4+1];
    float ox2 = boxes[idx*4+2] + off, oy2 = boxes[idx*4+3];
    boff[idx*4+0] = ox1; boff[idx*4+1] = oy1;
    boff[idx*4+2] = ox2; boff[idx*4+3] = oy2;
    area[idx] = (ox2 - ox1) * (oy2 - oy1);
  }
  __syncthreads();

  for (int k = 0; k < MAX_DET; ++k) {
    float bs = -1e30f; int bi = -1;
    for (int idx = tid; idx < NCB; idx += 256)
      if (!rem[idx]) {
        float s = sc[idx];
        if (s > bs || (s == bs && (bi < 0 || idx < bi))) { bs = s; bi = idx; }
      }
    s_sc[tid] = bs; s_si[tid] = bi; __syncthreads();
    for (int o = 128; o > 0; o >>= 1) {
      if (tid < o) {
        float so = s_sc[tid + o]; int io = s_si[tid + o];
        if (io >= 0 && (so > s_sc[tid] ||
            (so == s_sc[tid] && (s_si[tid] < 0 || io < s_si[tid])))) {
          s_sc[tid] = so; s_si[tid] = io;
        }
      }
      __syncthreads();
    }
    int best = s_si[0];
    if (tid == 0) {
      if (best >= 0) {
        out[((size_t)i * MAX_DET + k) * 4 + 0] = boxes[best*4+0];
        out[((size_t)i * MAX_DET + k) * 4 + 1] = boxes[best*4+1];
        out[((size_t)i * MAX_DET + k) * 4 + 2] = boxes[best*4+2];
        out[((size_t)i * MAX_DET + k) * 4 + 3] = boxes[best*4+3];
        out[BB_*MAX_DET*4 + (size_t)i * MAX_DET + k] = sc[best];
        out[BB_*MAX_DET*5 + (size_t)i * MAX_DET + k] = (float)lab[best];
        rem[best] = 1;
      } else {
        out[((size_t)i * MAX_DET + k) * 4 + 0] = 0.f;
        out[((size_t)i * MAX_DET + k) * 4 + 1] = 0.f;
        out[((size_t)i * MAX_DET + k) * 4 + 2] = 0.f;
        out[((size_t)i * MAX_DET + k) * 4 + 3] = 0.f;
        out[BB_*MAX_DET*4 + (size_t)i * MAX_DET + k] = 0.f;
        out[BB_*MAX_DET*5 + (size_t)i * MAX_DET + k] = 0.f;
      }
    }
    __syncthreads();
    if (best >= 0) {
      float bx1 = boff[best*4+0], by1 = boff[best*4+1];
      float bx2 = boff[best*4+2], by2 = boff[best*4+3];
      float ab = area[best];
      for (int idx = tid; idx < NCB; idx += 256) {
        if (!rem[idx]) {
          float xi1 = fmaxf(bx1, boff[idx*4+0]);
          float yi1 = fmaxf(by1, boff[idx*4+1]);
          float xi2 = fminf(bx2, boff[idx*4+2]);
          float yi2 = fminf(by2, boff[idx*4+3]);
          float inter = fmaxf(xi2 - xi1, 0.f) * fmaxf(yi2 - yi1, 0.f);
          float iou = inter / (ab + area[idx] - inter + 1e-9f);
          if (iou > 0.5f) rem[idx] = 1;
        }
      }
    }
    __syncthreads();
  }
}

// ============================================================
// launch
// ============================================================
#define KSPLIT1 8   // FC1: K=12544 -> 8 x 1568 (1568 % 32 == 0)
#define KSPLIT2 4   // FC2: K=1024  -> 4 x 256

extern "C" void kernel_launch(void* const* d_in, const int* in_sizes, int n_in,
                              void* d_out, int out_size, void* d_ws, size_t ws_size,
                              hipStream_t stream) {
  const float* features  = (const float*)d_in[0];
  const float* proposals = (const float*)d_in[1];
  const int*   img_sizes = (const int*)d_in[2];
  const float* W1 = (const float*)d_in[3];
  const float* b1 = (const float*)d_in[4];
  const float* W2 = (const float*)d_in[5];
  const float* b2 = (const float*)d_in[6];
  const float* Wc = (const float*)d_in[7];
  const float* bc = (const float*)d_in[8];
  const float* Wb = (const float*)d_in[9];
  const float* bb = (const float*)d_in[10];
  float* out = (float*)d_out;
  float* ws  = (float*)d_ws;

  float* feats = ws;                                   // 256*12544
  float* x1    = feats + (size_t)NROI * FEAT_DIM;      // 256*1024
  float* x2    = x1 + (size_t)NROI * HID;              // 256*1024
  float* probs = x2 + (size_t)NROI * HID;              // 256*21
  float* reg   = probs + (size_t)NROI * NUM_CLASSES;   // 256*84
  float* post  = reg + (size_t)NROI * NUM_CLASSES * 4; // 2*30720
  float* part  = post + (size_t)BB_ * 30720;           // KSPLIT1*256*1024

  roi_align_kernel<<<NROI, 256, 0, stream>>>(features, proposals, feats);

  // FC1: 256 blocks for HBM-level MLP; reduce applies bias+ReLU
  gemm_bf16x3_splitk_kernel<<<dim3(HID / TM, NROI / TN, KSPLIT1), 256, 0, stream>>>(
      feats, W1, part, FEAT_DIM, HID, FEAT_DIM / KSPLIT1);
  reduce_bias_relu_kernel<<<(NROI * HID / 4) / 256, 256, 0, stream>>>(
      part, b1, x1, HID, KSPLIT1);

  // FC2
  gemm_bf16x3_splitk_kernel<<<dim3(HID / TM, NROI / TN, KSPLIT2), 256, 0, stream>>>(
      x1, W2, part, HID, HID, HID / KSPLIT2);
  reduce_bias_relu_kernel<<<(NROI * HID / 4) / 256, 256, 0, stream>>>(
      part, b2, x2, HID, KSPLIT2);

  heads_kernel<<<NROI, 128, 0, stream>>>(x2, Wc, bc, Wb, bb, probs, reg);

  post_kernel<<<BB_, 256, 0, stream>>>(probs, reg, proposals, img_sizes, post, out);
}
